// FlowMatchingActionHeadIDM_25580825215576
// MI455X (gfx1250) — compile-verified, matrix operand order checked by reading, not run on hardware
//
#include <hip/hip_runtime.h>
#include <hip/hip_bf16.h>

// ---------------------------------------------------------------------------
// FlowMatchingActionHead IDM chain on MI455X (gfx1250, wave32, WMMA bf16)
//   B=64, T=32, ACTION_DIM=32, HIDDEN=1024, NUM_EMB=32
// Category-gathered GEMM chain on v_wmma_f32_16x16x32_bf16.
//  - fp32 weights streamed from HBM (unavoidable ~1GB term @ 23.3 TB/s)
//  - activations staged per-block in LDS as bf16 (double-buffered), read back
//    via ds_load_b128 fragments -> 4x less A traffic, shared conversions
//  - fp32 accumulate, bias + swish/relu epilogue in fp32
// ---------------------------------------------------------------------------

#define BB 64
#define TT 32
#define HID 1024
#define LDS_STRIDE 40   // bf16 per LDS row (80B): 16B-aligned, bank-friendly

typedef __attribute__((ext_vector_type(16))) __bf16 v16bf;
typedef __attribute__((ext_vector_type(8)))  __bf16 v8bf;
typedef __attribute__((ext_vector_type(8)))  float  v8f;

// ---------------------------------------------------------------------------
// tau kernel: x1[b, t, 1024 + j] = sin/cos(timesteps[b] * exp(-j'*ln(1e4)/512))
// ---------------------------------------------------------------------------
__global__ void tau_kernel(const int* __restrict__ timesteps,
                           float* __restrict__ x1) {
    const int b = blockIdx.x;
    const float t = (float)timesteps[b];
    const float coef = 9.210340371976184f / 512.0f;  // ln(10000)/half
    for (int j = threadIdx.x; j < HID; j += blockDim.x) {
        const int hidx = j & 511;
        const float f = t * __expf(-(float)hidx * coef);
        const float val = (j < 512) ? __sinf(f) : __cosf(f);
        float* p = x1 + ((size_t)b * TT) * (2 * HID) + HID + j;
#pragma unroll
        for (int tt = 0; tt < TT; ++tt) p[(size_t)tt * (2 * HID)] = val;
    }
}

// ---------------------------------------------------------------------------
// Y[b] = act( X[b](32 x K) @ W[cat[b]](K x N) + bias )
// EPI: 0 = identity, 1 = swish, 2 = relu.  N compile-time (immediate offsets).
// Block = 2*nwdim waves; wave (mw,nw) owns 16x16 tile at (mw*16, n0+nw*16).
// ---------------------------------------------------------------------------
template <int EPI, int N>
__global__ void cat_gemm(const float* __restrict__ X,
                         const float* __restrict__ W,
                         const float* __restrict__ Bias,
                         const int* __restrict__ cat_ids,
                         float* __restrict__ Y,
                         int K, int ldY) {
    __shared__ __bf16 sA[2][32 * LDS_STRIDE];   // double-buffered 32x32 bf16

    const int waves  = blockDim.x >> 5;
    const int nwdim  = waves >> 1;          // waves along N
    const int tileW  = nwdim * 16;          // block tile width in N
    const int ntiles = N / tileW;

    const int b    = blockIdx.x / ntiles;
    const int n0   = (blockIdx.x % ntiles) * tileW;
    const int wave = threadIdx.x >> 5;
    const int lane = threadIdx.x & 31;
    const int mw   = wave / nwdim;          // 0..1
    const int nw   = wave % nwdim;          // 0..nwdim-1
    const int m0   = mw * 16;
    const int nn   = n0 + nw * 16;
    const int half = lane >> 4;             // lane group 0/1
    const int lid  = lane & 15;

    const int c = cat_ids[b];
    const float* __restrict__ Xb   = X + (size_t)b * TT * (size_t)K;
    const float* __restrict__ Bcol = W + (size_t)c * K * N + (nn + lid);

    // This lane's A-fragment row in LDS (rows m0+lid; halves differ in k only)
    const __bf16* sArow = &sA[0][0] + (m0 + lid) * LDS_STRIDE + 8 * half;

    v8f acc = {};
    int buf = 0;
    for (int k0 = 0; k0 < K; k0 += 32, buf ^= 1) {
        // ---- cooperative A chunk: 32 rows x 32 k, fp32 -> bf16 into LDS ----
        for (int u = threadIdx.x; u < 256; u += blockDim.x) {
            const int r  = u >> 3;          // row 0..31
            const int kc = (u & 7) << 2;    // k col 0,4,..,28
            const float4 av = *(const float4*)(Xb + (size_t)r * K + k0 + kc);
            __bf16* dst = &sA[buf][r * LDS_STRIDE + kc];
            dst[0] = (__bf16)av.x; dst[1] = (__bf16)av.y;
            dst[2] = (__bf16)av.z; dst[3] = (__bf16)av.w;
        }

        // ---- B fragment global loads (issued before barrier to overlap) ----
        const float* bp = Bcol + (size_t)(k0 + half * 8) * N;
        float bv[16];
#pragma unroll
        for (int j = 0; j < 8; ++j) {
            bv[j]     = bp[(size_t)j * N];          // k rel 0..7
            bv[j + 8] = bp[(size_t)(16 + j) * N];   // k rel 16..23
        }

        __syncthreads();

        // ---- A fragment: two ds_load_b128 of 8 bf16 each ----
        const __bf16* sp = sArow + buf * (32 * LDS_STRIDE);
        const v8bf alo = *(const v8bf*)sp;          // k rel 8h .. 8h+7
        const v8bf ahi = *(const v8bf*)(sp + 16);   // k rel 16+8h .. 23+8h
        const v16bf af = __builtin_shufflevector(
            alo, ahi, 0, 1, 2, 3, 4, 5, 6, 7, 8, 9, 10, 11, 12, 13, 14, 15);

        // ---- B fragment cvt ----
        v16bf bfr;
#pragma unroll
        for (int j = 0; j < 16; ++j) bfr[j] = (__bf16)bv[j];

        acc = __builtin_amdgcn_wmma_f32_16x16x32_bf16(
            /*neg_a=*/false, af, /*neg_b=*/false, bfr,
            /*c_mod=*/(short)0, acc, /*reuse_a=*/false, /*reuse_b=*/false);
    }

    // ---- epilogue: C/D layout -> row m0 + half*8 + v, col nn + lid ----
    const float bias = Bias[(size_t)c * N + nn + lid];
    float* __restrict__ yp =
        Y + ((size_t)b * TT + m0 + half * 8) * (size_t)ldY + nn + lid;
#pragma unroll
    for (int v = 0; v < 8; ++v) {
        float val = acc[v] + bias;
        if (EPI == 1) val = val * (1.0f / (1.0f + __expf(-val)));   // swish
        if (EPI == 2) val = fmaxf(val, 0.0f);                       // relu
        yp[(size_t)v * ldY] = val;
    }
}

// ---------------------------------------------------------------------------
extern "C" void kernel_launch(void* const* d_in, const int* in_sizes, int n_in,
                              void* d_out, int out_size, void* d_ws, size_t ws_size,
                              hipStream_t stream) {
    (void)in_sizes; (void)n_in; (void)out_size; (void)ws_size;

    const float* actions   = (const float*)d_in[0];
    const int*   timesteps = (const int*)d_in[1];
    const int*   cat_ids   = (const int*)d_in[2];
    const float* W1  = (const float*)d_in[3];
    const float* b1  = (const float*)d_in[4];
    const float* W2  = (const float*)d_in[5];
    const float* b2  = (const float*)d_in[6];
    const float* W3  = (const float*)d_in[7];
    const float* b3  = (const float*)d_in[8];
    const float* dW1 = (const float*)d_in[9];
    const float* db1 = (const float*)d_in[10];
    const float* dW2 = (const float*)d_in[11];
    const float* db2 = (const float*)d_in[12];
    float* out = (float*)d_out;

    float* x1 = (float*)d_ws;                       // (B, 32, 2048)  16 MB
    float* x2 = x1 + (size_t)BB * TT * 2 * HID;     // (B, 32, 1024)   8 MB
    float* x3 = x2 + (size_t)BB * TT * HID;         // (B, 32, 1024)   8 MB
    float* h  = x3 + (size_t)BB * TT * HID;         // (B, 32, 1024)   8 MB

    // tau half of x1
    tau_kernel<<<BB, 256, 0, stream>>>(timesteps, x1);
    // L1: actions(32x32) @ W1 -> x1[:, :, 0:1024] (row stride 2048)
    cat_gemm<0, HID><<<BB * (HID / 64), 256, 0, stream>>>(
        actions, W1, b1, cat_ids, x1, /*K=*/32, /*ldY=*/2 * HID);
    // L2 + swish: x1(32x2048) @ W2 -> x2
    cat_gemm<1, HID><<<BB * (HID / 64), 256, 0, stream>>>(
        x1, W2, b2, cat_ids, x2, /*K=*/2 * HID, /*ldY=*/HID);
    // L3: x2 @ W3 -> x3
    cat_gemm<0, HID><<<BB * (HID / 64), 256, 0, stream>>>(
        x2, W3, b3, cat_ids, x3, /*K=*/HID, /*ldY=*/HID);
    // L4 + relu: x3 @ dW1 -> h
    cat_gemm<2, HID><<<BB * (HID / 64), 256, 0, stream>>>(
        x3, dW1, db1, cat_ids, h, /*K=*/HID, /*ldY=*/HID);
    // L5: h @ dW2 -> out (N=32 -> 4 waves, 2x2 tiles)
    cat_gemm<0, 32><<<BB, 128, 0, stream>>>(
        h, dW2, db2, cat_ids, out, /*K=*/HID, /*ldY=*/32);
}
